// AdaptiveMixingAttention_6219112645321
// MI455X (gfx1250) — matrix operand order repaired and checked by reading, not compiled, not run on hardware
//
#include <hip/hip_runtime.h>
#include <hip/hip_bf16.h>

// AdaMixer adaptive mixing for gfx1250 (wave32, WMMA bf16, f32 accum).
//  K1: params = query @ Wg + bg          -> ws (bf16)   [128x128 tile, 8 acc/wave]
//  K2: per (n,g): relu(ln(x@M)); relu(ln(S@..)) -> ws (bf16)
//  K3: partial[z] = Y @ Wo (split-K x8)  -> ws (f32)    [128x128 tile, 8 acc/wave]
//  K4: out = sum_z partial + bo + query  -> d_out (f32)

typedef __bf16 bf16;
typedef __attribute__((ext_vector_type(16))) __bf16 v16bf;
typedef __attribute__((ext_vector_type(8)))  float  v8f;

#define BNQ 2048      // B*N
#define DQ  256       // query dim
#define GT  32768     // G*T
#define TPG 8192      // T per group
#define MP  4096      // C*C
#define PP  32        // in_points
#define SPLITK 8      // K split for output projection (32768/8 = 4096 each)
#define LNEPS 1e-5f

// ---- WMMA fragment loaders (CDNA5 ISA §7.12.2 layouts, wave32) ----
// A (16x32 bf16, MxK): lane&15 = row; lane>>4 = K-half within 4-VGPR group.
__device__ __forceinline__ v16bf load_frag_a(const bf16* s, int ld, int row0,
                                             int col0, int lane) {
  v16bf a;
  const int m  = row0 + (lane & 15);
  const int kh = (lane >> 4) << 3;
#pragma unroll
  for (int v = 0; v < 8; ++v) {
    const int kb = col0 + ((v >> 2) << 4) + kh + ((v & 3) << 1);
    a[2 * v]     = s[m * ld + kb];
    a[2 * v + 1] = s[m * ld + kb + 1];
  }
  return a;
}

// B (32x16 bf16, KxN): lane&15 = col; K along the vector in A's (table) order.
__device__ __forceinline__ v16bf load_frag_b(const bf16* s, int ld, int row0,
                                             int col0, int lane) {
  v16bf b;
  const int nc = col0 + (lane & 15);
  const int kh = row0 + ((lane >> 4) << 3);
#pragma unroll
  for (int e = 0; e < 16; ++e) b[e] = s[(kh + (e & 7) + ((e >> 3) << 4)) * ld + nc];
  return b;
}

// C/D f32 16x16: acc[r] holds (m = 8*(lane>>4) + r, n = lane&15).

__device__ __forceinline__ float wave_sum(float v) {
#pragma unroll
  for (int o = 16; o > 0; o >>= 1) v += __shfl_down(v, o, 32);
  return v;
}

// -------------------- Kernel 1: params = query @ Wg + bg --------------------
// grid (16, 256), block 256 (8 waves). Block tile 128x128.
// Wave w: rows (w&3)*32 (2 A-frags), cols (w>>2)*64 (4 B-frags) -> 8 WMMA/chunk.
__global__ __launch_bounds__(256) void params_gemm_kernel(
    const float* __restrict__ Q, const float* __restrict__ Wg,
    const float* __restrict__ bg, bf16* __restrict__ Pout) {
  __shared__ bf16 sQ[128][32];   // 8 KB
  __shared__ bf16 sW[32][128];   // 8 KB
  const int tid = threadIdx.x, lane = tid & 31, w = tid >> 5;
  const int rb = (w & 3) * 32, nb = (w >> 2) * 64;
  const int row0 = blockIdx.x * 128;
  const int col0 = blockIdx.y * 128;
  v8f acc[2][4] = {};
  for (int kk = 0; kk < DQ; kk += 32) {
    {  // stage query tile 128x32 (f32 -> bf16): 16 elems/thread
      const int r = tid >> 1, c = (tid & 1) << 4;
      const float* src = Q + (size_t)(row0 + r) * DQ + kk + c;
#pragma unroll
      for (int q = 0; q < 4; ++q) {
        const float4 v = *(const float4*)(src + 4 * q);
        sQ[r][c + 4 * q]     = (bf16)v.x; sQ[r][c + 4 * q + 1] = (bf16)v.y;
        sQ[r][c + 4 * q + 2] = (bf16)v.z; sQ[r][c + 4 * q + 3] = (bf16)v.w;
      }
    }
    {  // stage Wg tile 32x128 (f32 -> bf16): 16 elems/thread
      const int r = tid >> 3, c = (tid & 7) << 4;
      const float* src = Wg + (size_t)(kk + r) * GT + col0 + c;
#pragma unroll
      for (int q = 0; q < 4; ++q) {
        const float4 v = *(const float4*)(src + 4 * q);
        sW[r][c + 4 * q]     = (bf16)v.x; sW[r][c + 4 * q + 1] = (bf16)v.y;
        sW[r][c + 4 * q + 2] = (bf16)v.z; sW[r][c + 4 * q + 3] = (bf16)v.w;
      }
      if (kk + 32 < DQ)  // global_prefetch_b8 for next K chunk
        __builtin_prefetch(src + (size_t)32 * GT, 0, 1);
    }
    __syncthreads();
    v16bf a0 = load_frag_a(&sQ[0][0], 32, rb, 0, lane);
    v16bf a1 = load_frag_a(&sQ[0][0], 32, rb + 16, 0, lane);
#pragma unroll
    for (int j = 0; j < 4; ++j) {
      const v16bf b = load_frag_b(&sW[0][0], 128, 0, nb + j * 16, lane);
      acc[0][j] = __builtin_amdgcn_wmma_f32_16x16x32_bf16(
          false, a0, false, b, (short)0, acc[0][j], false, false);
      acc[1][j] = __builtin_amdgcn_wmma_f32_16x16x32_bf16(
          false, a1, false, b, (short)0, acc[1][j], false, false);
    }
    __syncthreads();
  }
#pragma unroll
  for (int ii = 0; ii < 2; ++ii) {
    const int mb = row0 + rb + ii * 16 + ((lane >> 4) << 3);
#pragma unroll
    for (int j = 0; j < 4; ++j) {
      const int nn = col0 + nb + j * 16 + (lane & 15);
      const float bgv = bg[nn];
#pragma unroll
      for (int r = 0; r < 8; ++r)
        Pout[(size_t)(mb + r) * GT + nn] = (bf16)(acc[ii][j][r] + bgv);
    }
  }
}

// ------------- Kernel 2: per-(n,g) channel mix + LN + spatial mix -----------
// grid (4, 2048), block 256 (8 waves). ~60 KB static LDS.
__global__ __launch_bounds__(256) void mixer_kernel(
    const float* __restrict__ X, const bf16* __restrict__ Pm,
    bf16* __restrict__ Y) {
  __shared__ bf16  sM[64][64];     // channel mixing matrix M
  __shared__ bf16  sS[128][32];    // spatial mixing matrix S
  __shared__ float sO1[32][64];    // x@M   (f32, for LN)
  __shared__ float sO2[128][64];   // S@o1  (f32, for LN)
  __shared__ bf16  sXb[32][64];    // x bf16, reused as relu(ln(o1))
  __shared__ float sred[2][8];
  __shared__ float sStat[2];

  const int g = blockIdx.x, n = blockIdx.y;
  const int tid = threadIdx.x, lane = tid & 31, w = tid >> 5;
  const float* xg = X + ((size_t)n * 4 + g) * (PP * 64);
  const bf16*  pg = Pm + (size_t)n * GT + (size_t)g * TPG;

#pragma unroll
  for (int i = 0; i < 8; ++i) {  // x tile 32x64 -> bf16
    const int idx = i * 256 + tid;
    sXb[idx >> 6][idx & 63] = (bf16)xg[idx];
  }
#pragma unroll
  for (int i = 0; i < 16; ++i) {  // M 64x64
    const int idx = i * 256 + tid;
    sM[idx >> 6][idx & 63] = pg[idx];
  }
#pragma unroll
  for (int i = 0; i < 16; ++i) {  // S 128x32
    const int idx = i * 256 + tid;
    sS[idx >> 5][idx & 31] = pg[MP + idx];
  }
  __syncthreads();

  // ---- out1 = x @ M : 32x64, K=64. 8 waves x one 16x16 tile. ----
  {
    const int mi = w & 1, ni = w >> 1;
    v8f acc = {};
#pragma unroll
    for (int kc = 0; kc < 64; kc += 32) {
      const v16bf a = load_frag_a(&sXb[0][0], 64, mi * 16, kc, lane);
      const v16bf b = load_frag_b(&sM[0][0], 64, kc, ni * 16, lane);
      acc = __builtin_amdgcn_wmma_f32_16x16x32_bf16(
          false, a, false, b, (short)0, acc, false, false);
    }
    const int nn = ni * 16 + (lane & 15);
    const int mb = mi * 16 + ((lane >> 4) << 3);
#pragma unroll
    for (int r = 0; r < 8; ++r) sO1[mb + r][nn] = acc[r];
  }
  __syncthreads();

  // ---- LN over 32x64 + ReLU -> bf16 into sXb ----
  {
    float s = 0.f, s2 = 0.f;
#pragma unroll
    for (int i = 0; i < 8; ++i) {
      const int idx = i * 256 + tid;
      const float v = sO1[idx >> 6][idx & 63];
      s += v; s2 += v * v;
    }
    s = wave_sum(s); s2 = wave_sum(s2);
    if (lane == 0) { sred[0][w] = s; sred[1][w] = s2; }
    __syncthreads();
    if (tid == 0) {
      float a = 0.f, b = 0.f;
#pragma unroll
      for (int i = 0; i < 8; ++i) { a += sred[0][i]; b += sred[1][i]; }
      const float mu = a * (1.f / 2048.f);
      const float var = b * (1.f / 2048.f) - mu * mu;
      sStat[0] = mu; sStat[1] = rsqrtf(var + LNEPS);
    }
    __syncthreads();
    const float mu = sStat[0], rs = sStat[1];
#pragma unroll
    for (int i = 0; i < 8; ++i) {
      const int idx = i * 256 + tid;
      float v = (sO1[idx >> 6][idx & 63] - mu) * rs;
      sXb[idx >> 6][idx & 63] = (bf16)fmaxf(v, 0.f);
    }
  }
  __syncthreads();

  // ---- out2 = S @ out1 : 128x64, K=32. A-frag reused across 4 col tiles. --
  {
    const v16bf a = load_frag_a(&sS[0][0], 32, w * 16, 0, lane);
    const int nlane = lane & 15;
    const int mb = w * 16 + ((lane >> 4) << 3);
#pragma unroll
    for (int ni2 = 0; ni2 < 4; ++ni2) {
      const v16bf b = load_frag_b(&sXb[0][0], 64, 0, ni2 * 16, lane);
      v8f acc = {};
      acc = __builtin_amdgcn_wmma_f32_16x16x32_bf16(
          false, a, false, b, (short)0, acc, false, false);
#pragma unroll
      for (int r = 0; r < 8; ++r) sO2[mb + r][ni2 * 16 + nlane] = acc[r];
    }
  }
  __syncthreads();

  // ---- LN over 128x64 + ReLU -> Y (bf16) ----
  {
    float s = 0.f, s2 = 0.f;
#pragma unroll
    for (int i = 0; i < 32; ++i) {
      const int idx = i * 256 + tid;
      const float v = sO2[idx >> 6][idx & 63];
      s += v; s2 += v * v;
    }
    s = wave_sum(s); s2 = wave_sum(s2);
    if (lane == 0) { sred[0][w] = s; sred[1][w] = s2; }
    __syncthreads();
    if (tid == 0) {
      float a = 0.f, b = 0.f;
#pragma unroll
      for (int i = 0; i < 8; ++i) { a += sred[0][i]; b += sred[1][i]; }
      const float mu = a * (1.f / 8192.f);
      const float var = b * (1.f / 8192.f) - mu * mu;
      sStat[0] = mu; sStat[1] = rsqrtf(var + LNEPS);
    }
    __syncthreads();
    const float mu = sStat[0], rs = sStat[1];
    bf16* yg = Y + (size_t)n * GT + (size_t)g * TPG;
#pragma unroll
    for (int i = 0; i < 32; ++i) {
      const int idx = i * 256 + tid;
      float v = (sO2[idx >> 6][idx & 63] - mu) * rs;
      yg[idx] = (bf16)fmaxf(v, 0.f);
    }
  }
}

// ------------- Kernel 3: partial[z] = Y @ Wo over K segment z ---------------
// grid (16, 2, 8), block 256. Block tile 128x128, K segment = 4096.
__global__ __launch_bounds__(256) void out_gemm_kernel(
    const bf16* __restrict__ Y, const float* __restrict__ Wo,
    float* __restrict__ partial) {
  __shared__ bf16 sA[128][32];   // 8 KB (Y is already bf16)
  __shared__ bf16 sB[32][128];   // 8 KB
  const int tid = threadIdx.x, lane = tid & 31, w = tid >> 5;
  const int rb = (w & 3) * 32, nb = (w >> 2) * 64;
  const int row0 = blockIdx.x * 128;
  const int col0 = blockIdx.y * 128;
  const int z = blockIdx.z;
  const int k0 = z * (GT / SPLITK), k1 = k0 + (GT / SPLITK);
  v8f acc[2][4] = {};
  for (int kk = k0; kk < k1; kk += 32) {
    {  // stage Y tile 128x32: 16 bf16 (32 B) per thread, raw copy
      const int r = tid >> 1, c = (tid & 1) << 4;
      const bf16* src = Y + (size_t)(row0 + r) * GT + kk + c;
      *(uint4*)&sA[r][c]     = *(const uint4*)src;
      *(uint4*)&sA[r][c + 8] = *(const uint4*)(src + 8);
    }
    {  // stage Wo tile 32x128 (f32 -> bf16): 16 elems/thread
      const int r = tid >> 3, c = (tid & 7) << 4;
      const float* src = Wo + (size_t)(kk + r) * DQ + col0 + c;
#pragma unroll
      for (int q = 0; q < 4; ++q) {
        const float4 v = *(const float4*)(src + 4 * q);
        sB[r][c + 4 * q]     = (bf16)v.x; sB[r][c + 4 * q + 1] = (bf16)v.y;
        sB[r][c + 4 * q + 2] = (bf16)v.z; sB[r][c + 4 * q + 3] = (bf16)v.w;
      }
      if (kk + 32 < k1)
        __builtin_prefetch(src + (size_t)32 * DQ, 0, 1);
    }
    __syncthreads();
    v16bf a0 = load_frag_a(&sA[0][0], 32, rb, 0, lane);
    v16bf a1 = load_frag_a(&sA[0][0], 32, rb + 16, 0, lane);
#pragma unroll
    for (int j = 0; j < 4; ++j) {
      const v16bf b = load_frag_b(&sB[0][0], 128, 0, nb + j * 16, lane);
      acc[0][j] = __builtin_amdgcn_wmma_f32_16x16x32_bf16(
          false, a0, false, b, (short)0, acc[0][j], false, false);
      acc[1][j] = __builtin_amdgcn_wmma_f32_16x16x32_bf16(
          false, a1, false, b, (short)0, acc[1][j], false, false);
    }
    __syncthreads();
  }
  float* pz = partial + (size_t)z * BNQ * DQ;
#pragma unroll
  for (int ii = 0; ii < 2; ++ii) {
    const int mb = row0 + rb + ii * 16 + ((lane >> 4) << 3);
#pragma unroll
    for (int j = 0; j < 4; ++j) {
      const int nn = col0 + nb + j * 16 + (lane & 15);
#pragma unroll
      for (int r = 0; r < 8; ++r)
        pz[(size_t)(mb + r) * DQ + nn] = acc[ii][j][r];
    }
  }
}

// ------------- Kernel 4: out = sum_z partial + bo + query -------------------
__global__ __launch_bounds__(256) void reduce_kernel(
    const float* __restrict__ partial, const float* __restrict__ bo,
    const float* __restrict__ Q, float* __restrict__ out) {
  const int idx = blockIdx.x * 256 + threadIdx.x;
  const int d = idx & (DQ - 1);
  float s = bo[d] + Q[idx];
#pragma unroll
  for (int z = 0; z < SPLITK; ++z)
    s += partial[(size_t)z * BNQ * DQ + idx];
  out[idx] = s;
}

extern "C" void kernel_launch(void* const* d_in, const int* in_sizes, int n_in,
                              void* d_out, int out_size, void* d_ws,
                              size_t ws_size, hipStream_t stream) {
  const float* x     = (const float*)d_in[0];
  const float* query = (const float*)d_in[1];
  const float* Wg    = (const float*)d_in[2];
  const float* bg    = (const float*)d_in[3];
  const float* Wo    = (const float*)d_in[4];
  const float* bo    = (const float*)d_in[5];
  float* out = (float*)d_out;

  bf16* params = (bf16*)d_ws;                                   // 128 MiB
  bf16* Y = (bf16*)((char*)d_ws + (size_t)BNQ * GT * 2);        // 128 MiB
  float* partial = (float*)((char*)d_ws + (size_t)2 * BNQ * GT * 2);  // 16 MiB

  params_gemm_kernel<<<dim3(BNQ / 128, GT / 128), 256, 0, stream>>>(query, Wg,
                                                                    bg, params);
  mixer_kernel<<<dim3(4, BNQ), 256, 0, stream>>>(x, params, Y);
  out_gemm_kernel<<<dim3(BNQ / 128, DQ / 128, SPLITK), 256, 0, stream>>>(
      Y, Wo, partial);
  reduce_kernel<<<dim3(BNQ * DQ / 256), 256, 0, stream>>>(partial, bo, query,
                                                          out);
}
